// HyenaBaseBlock_25718264168566
// MI455X (gfx1250) — compile-verified, matrix operand order checked by reading, not run on hardware
//
#include <hip/hip_runtime.h>
#include <hip/hip_bf16.h>
#include <cstdint>
#include <math.h>

// ---------------------------------------------------------------------------
// Problem constants (HyenaBaseBlock): B=4, L=4096, D=1024, P=128
// ---------------------------------------------------------------------------
constexpr int kB  = 4;
constexpr int kL  = 4096;
constexpr int kD  = 1024;
constexpr int kP  = 128;
constexpr int kNF = 8192;          // FFT length n = 2L
constexpr int kH1 = kP + kD;       // 1152
constexpr int kDF = 2 * kD;        // 2048
constexpr int kMT = kB * kL;       // 16384 tokens

typedef __attribute__((ext_vector_type(16))) __bf16   v16bf;
typedef __attribute__((ext_vector_type(8)))  float    v8f;
typedef __attribute__((ext_vector_type(4)))  uint32_t v4u;
typedef __attribute__((ext_vector_type(8)))  int32_t  v8i;
typedef __attribute__((ext_vector_type(4)))  int32_t  v4i;

#define WMMA_BF16(a, b, c) \
  __builtin_amdgcn_wmma_f32_16x16x32_bf16(false, (a), false, (b), (short)0, (c), false, false)

// Tensor Data Mover availability (device pass only; host pass parses fallback)
#if defined(__has_builtin)
#if __has_builtin(__builtin_amdgcn_tensor_load_to_lds)
#define HAS_TDM 1
#endif
#endif
#ifndef HAS_TDM
#define HAS_TDM 0
#endif

// ---------------------------------------------------------------------------
// f32 -> bf16 conversion (plain, and transposing for weights -> [N][K])
// ---------------------------------------------------------------------------
__global__ __launch_bounds__(256) void cvt_bf16_kernel(const float* __restrict__ in,
                                                       __bf16* __restrict__ out, int n) {
  int i = blockIdx.x * 256 + threadIdx.x;
  if (i < n) out[i] = (__bf16)in[i];
}

// in: [K][N] f32 row-major; out: [N][K] bf16 row-major
__global__ __launch_bounds__(256) void cvt_bf16_t_kernel(const float* __restrict__ in,
                                                         __bf16* __restrict__ out,
                                                         int K, int N) {
  int i = blockIdx.x * 256 + threadIdx.x;
  if (i < K * N) {
    int n = i / K, k = i % K;
    out[i] = (__bf16)in[(size_t)k * N + n];
  }
}

// ---------------------------------------------------------------------------
// LayerNorm over rows of D (=1024), optional fused elementwise multiply
// (y = x1 * cwhz) with pre-LN product emitted for the later residual.
// ---------------------------------------------------------------------------
__global__ __launch_bounds__(256) void ln_kernel(
    const float* __restrict__ X, const float* __restrict__ X2,
    const float* __restrict__ g, const float* __restrict__ bta,
    float* __restrict__ outF, __bf16* __restrict__ outB,
    float* __restrict__ outProd, int D)
{
  __shared__ float red[256][2];
  const int row = blockIdx.x;
  const int tid = threadIdx.x;
  const int per = D >> 8;
  const size_t base = (size_t)row * D;

  float v[8];
  float s = 0.0f, sq = 0.0f;
  for (int i = 0; i < per; ++i) {
    int c = tid + (i << 8);
    float t = X[base + c];
    if (X2) t *= X2[base + c];
    if (outProd) outProd[base + c] = t;
    v[i] = t;
    s += t; sq += t * t;
  }
  red[tid][0] = s; red[tid][1] = sq;
  __syncthreads();
  for (int off = 128; off > 0; off >>= 1) {
    if (tid < off) {
      red[tid][0] += red[tid + off][0];
      red[tid][1] += red[tid + off][1];
    }
    __syncthreads();
  }
  const float mean = red[0][0] / (float)D;
  const float var  = red[0][1] / (float)D - mean * mean;
  const float rstd = rsqrtf(var + 1e-5f);
  for (int i = 0; i < per; ++i) {
    int c = tid + (i << 8);
    float y = (v[i] - mean) * rstd * g[c] + bta[c];
    if (outF) outF[base + c] = y;
    if (outB) outB[base + c] = (__bf16)y;
  }
}

// ---------------------------------------------------------------------------
// WMMA bf16 GEMM with TDM tile staging.
//   C[M,N] = A[M,K] @ Bt[N,K]^T + bias   (Bt is pre-transposed weights)
//   mode 0: SiLU -> bf16 out;  mode 1: f32 out + optional residual(s)
// Block tile 128x128, 8 waves in 4x2 grid, each wave 32x64 (2x4 fragments,
// 8 v_wmma per k-step). Tiles DMA'd by TDM into LDS with 1-DWORD row padding
// (row stride 34 elems) to spread fragment reads across banks; double
// buffered and pipelined on TENSORcnt.
// ---------------------------------------------------------------------------
constexpr int TMg = 128, TNg = 128, TKg = 32, SROW = 34;   // SROW: 32 + 2 pad

#if HAS_TDM
// 2D tile load: tile_dim0=32 (K), tile_dim1=128 (rows), data_size=2B,
// pad_enable=1, pad_interval=3 (16 DWORDs = one 64B row), pad_amount=0 (1 DWORD).
__device__ inline void tdm_load_tile(uint32_t lds_off, const __bf16* tile,
                                     uint32_t K_len, uint32_t rows_tot) {
  const uint64_t ga = (uint64_t)(uintptr_t)tile;
  v4u g0;
  g0.x = 1u;                                             // count=1 (valid user D#)
  g0.y = lds_off;                                        // LDS byte address
  g0.z = (uint32_t)(ga & 0xffffffffu);                   // global_addr[31:0]
  g0.w = (uint32_t)((ga >> 32) & 0x1ffffffu) | (2u << 30); // addr[56:32] | type=2
  v8i g1;
  g1[0] = (int)((1u << 16) | (1u << 20) | (3u << 22));   // data_size=2B, pad on
  g1[1] = (int)((K_len & 0xffffu) << 16);                // tensor_dim0 lo
  g1[2] = (int)(((K_len >> 16) & 0xffffu) | ((rows_tot & 0xffffu) << 16));
  g1[3] = (int)(((rows_tot >> 16) & 0xffffu) | ((uint32_t)TKg << 16)); // tile_dim0=32
  g1[4] = (int)TMg;                                      // tile_dim1=128, tile_dim2=0
  g1[5] = (int)K_len;                                    // tensor_dim0_stride lo
  g1[6] = 0;                                             // stride hi / dim1_stride lo
  g1[7] = 0;
  const v4i z4 = {0, 0, 0, 0};
#if __clang_major__ >= 23
  const v8i z8 = {0, 0, 0, 0, 0, 0, 0, 0};
  __builtin_amdgcn_tensor_load_to_lds(g0, g1, z4, z4, z8, 0);
#else
  __builtin_amdgcn_tensor_load_to_lds(g0, g1, z4, z4, 0);
#endif
}
#endif

__device__ inline v16bf load_frag(const __bf16* base, int lane) {
  union { v16bf v; uint32_t u[8]; } f;
  const __bf16* row = base + (size_t)(lane & 15) * SROW;
  const int g8 = (lane >> 4) * 8;
#pragma unroll
  for (int j = 0; j < 8; ++j) {
    int kp = ((j & 4) ? 16 : 0) + 2 * (j & 3) + g8;   // K-pair start (ISA layout)
    f.u[j] = *reinterpret_cast<const uint32_t*>(row + kp);
  }
  return f.v;
}

__global__ __launch_bounds__(256) void gemm_bf16_kernel(
    const __bf16* __restrict__ A, const __bf16* __restrict__ Bt,
    const float* __restrict__ bias, void* __restrict__ Cout,
    const float* __restrict__ res1, const float* __restrict__ res2,
    int M, int N, int K, int mode)
{
  __shared__ __bf16 As[2][TMg * SROW];
  __shared__ __bf16 Bs[2][TNg * SROW];

  const int tid  = threadIdx.x;
  const int lane = tid & 31;
  const int wave = tid >> 5;
  const int wm   = (wave >> 1) * 32;     // 4 wave-rows
  const int wn   = (wave & 1) * 64;      // 2 wave-cols
  const int m0   = blockIdx.y * TMg;
  const int n0   = blockIdx.x * TNg;
  const int nk   = K / TKg;

  v8f acc[2][4] = {};

#if HAS_TDM
  if (wave == 0) {
    tdm_load_tile((uint32_t)(uintptr_t)&As[0][0], A  + (size_t)m0 * K, K, M);
    tdm_load_tile((uint32_t)(uintptr_t)&Bs[0][0], Bt + (size_t)n0 * K, K, N);
  }
#endif

  for (int kt = 0; kt < nk; ++kt) {
    const int cur = kt & 1;
#if HAS_TDM
    if (wave == 0) {
      if (kt + 1 < nk) {
        const int kn = (kt + 1) * TKg;
        tdm_load_tile((uint32_t)(uintptr_t)&As[1 - cur][0], A  + (size_t)m0 * K + kn, K, M);
        tdm_load_tile((uint32_t)(uintptr_t)&Bs[1 - cur][0], Bt + (size_t)n0 * K + kn, K, N);
        __builtin_amdgcn_s_wait_tensorcnt((short)2);   // current tiles landed
      } else {
        __builtin_amdgcn_s_wait_tensorcnt((short)0);
      }
    }
    __syncthreads();
#else
    {
      const int kk0 = kt * TKg;
      for (int i = tid; i < TMg * TKg; i += 256) {
        int r = i >> 5, kk = i & 31;
        As[cur][r * SROW + kk] = A[(size_t)(m0 + r) * K + kk0 + kk];
      }
      for (int i = tid; i < TNg * TKg; i += 256) {
        int r = i >> 5, kk = i & 31;
        Bs[cur][r * SROW + kk] = Bt[(size_t)(n0 + r) * K + kk0 + kk];
      }
    }
    __syncthreads();
#endif

    v16bf af0 = load_frag(&As[cur][(wm)      * SROW], lane);
    v16bf af1 = load_frag(&As[cur][(wm + 16) * SROW], lane);
    v16bf bq0 = load_frag(&Bs[cur][(wn)      * SROW], lane);
    v16bf bq1 = load_frag(&Bs[cur][(wn + 16) * SROW], lane);
    v16bf bq2 = load_frag(&Bs[cur][(wn + 32) * SROW], lane);
    v16bf bq3 = load_frag(&Bs[cur][(wn + 48) * SROW], lane);

    acc[0][0] = WMMA_BF16(af0, bq0, acc[0][0]);
    acc[0][1] = WMMA_BF16(af0, bq1, acc[0][1]);
    acc[0][2] = WMMA_BF16(af0, bq2, acc[0][2]);
    acc[0][3] = WMMA_BF16(af0, bq3, acc[0][3]);
    acc[1][0] = WMMA_BF16(af1, bq0, acc[1][0]);
    acc[1][1] = WMMA_BF16(af1, bq1, acc[1][1]);
    acc[1][2] = WMMA_BF16(af1, bq2, acc[1][2]);
    acc[1][3] = WMMA_BF16(af1, bq3, acc[1][3]);
    __syncthreads();   // all waves done with buf[cur] before it is re-filled
  }

  const int gm = (lane >> 4) * 8;   // lanes 16-31 hold rows M=8..15 (ISA C/D layout)
  const int nl = lane & 15;
#pragma unroll
  for (int fi = 0; fi < 2; ++fi) {
#pragma unroll
    for (int fj = 0; fj < 4; ++fj) {
      const int mbase = m0 + wm + fi * 16 + gm;
      const int nn    = n0 + wn + fj * 16 + nl;
      const float bv  = bias ? bias[nn] : 0.0f;
#pragma unroll
      for (int r = 0; r < 8; ++r) {
        const size_t idx = (size_t)(mbase + r) * N + nn;
        float vv = acc[fi][fj][r] + bv;
        if (mode == 0) {
          float sg = vv / (1.0f + expf(-vv));          // SiLU
          ((__bf16*)Cout)[idx] = (__bf16)sg;
        } else {
          if (res1) vv += res1[idx];
          if (res2) vv += res2[idx];
          ((float*)Cout)[idx] = vv;
        }
      }
    }
  }
}

// ---------------------------------------------------------------------------
// Radix-2 Stockham FFT, N=8192, 256 threads, ping-pong in 128 KB LDS.
// ---------------------------------------------------------------------------
__device__ float2* fft8192(float2* src, float2* dst, float sign) {
  const int tid = threadIdx.x;
  for (int p = 1; p < kNF; p <<= 1) {
    __syncthreads();
    for (int i = 0; i < (kNF / 2) / 256; ++i) {
      int j = tid + (i << 8);
      int k = j & (p - 1);
      float2 u = src[j];
      float2 w = src[j + kNF / 2];
      float ang = sign * 3.14159265358979323846f * (float)k / (float)p;
      float sn, cs;
      __sincosf(ang, &sn, &cs);
      float2 wt = make_float2(w.x * cs - w.y * sn, w.x * sn + w.y * cs);
      int d0 = ((j - k) << 1) + k;
      dst[d0]     = make_float2(u.x + wt.x, u.y + wt.y);
      dst[d0 + p] = make_float2(u.x - wt.x, u.y - wt.y);
    }
    float2* t = src; src = dst; dst = t;
  }
  __syncthreads();
  return src;
}

__global__ __launch_bounds__(256) void fft_filter_kernel(
    const float* __restrict__ h, const float* __restrict__ a,
    float2* __restrict__ wfh)
{
  extern __shared__ float2 lds[];
  float2* X = lds;
  float2* Y = lds + kNF;
  const int d = blockIdx.x;
  const float decay = expf(a[0]);
  for (int i = threadIdx.x; i < kNF; i += 256) {
    float v = 0.0f;
    if (i < kL) v = expf(-(float)i * decay) * h[(size_t)i * kD + d];
    X[i] = make_float2(v, 0.0f);
  }
  float2* res = fft8192(X, Y, -1.0f);
  for (int i = threadIdx.x; i < kNF; i += 256)
    wfh[(size_t)d * kNF + i] = res[i];
}

__global__ __launch_bounds__(256) void fft_conv_kernel(
    const float* __restrict__ zn, const float2* __restrict__ wfh,
    float* __restrict__ cwhz)
{
  extern __shared__ float2 lds[];
  float2* X = lds;
  float2* Y = lds + kNF;
  const int bd = blockIdx.x;
  const int b  = bd / kD;
  const int d  = bd % kD;

  for (int i = threadIdx.x; i < kNF; i += 256) {
    float v = (i < kL) ? zn[((size_t)b * kL + i) * kD + d] : 0.0f;
    X[i] = make_float2(v, 0.0f);
  }
  float2* res = fft8192(X, Y, -1.0f);
  for (int i = threadIdx.x; i < kNF; i += 256) {
    float2 fz = res[i];
    float2 fw = wfh[(size_t)d * kNF + i];
    res[i] = make_float2(fz.x * fw.x - fz.y * fw.y,
                         fz.x * fw.y + fz.y * fw.x);
  }
  float2* other = (res == X) ? Y : X;
  float2* out = fft8192(res, other, 1.0f);
  const float invN = 1.0f / (float)kNF;
  for (int i = threadIdx.x; i < kL; i += 256)
    cwhz[((size_t)b * kL + i) * kD + d] = out[i].x * invN;
}

// ---------------------------------------------------------------------------
// Host orchestration
// ---------------------------------------------------------------------------
extern "C" void kernel_launch(void* const* d_in, const int* in_sizes, int n_in,
                              void* d_out, int out_size, void* d_ws, size_t ws_size,
                              hipStream_t stream)
{
  (void)in_sizes; (void)n_in; (void)out_size; (void)ws_size;

  const float* z       = (const float*)d_in[0];
  const float* x       = (const float*)d_in[1];
  const float* a       = (const float*)d_in[2];
  const float* pe      = (const float*)d_in[3];
  const float* lp1_w   = (const float*)d_in[4];
  const float* lp1_b   = (const float*)d_in[5];
  const float* lp2_w   = (const float*)d_in[6];
  const float* lp2_b   = (const float*)d_in[7];
  const float* ln_g    = (const float*)d_in[8];
  const float* ln_b    = (const float*)d_in[9];
  const float* f1_ln_g = (const float*)d_in[10];
  const float* f1_ln_b = (const float*)d_in[11];
  const float* f1_w1   = (const float*)d_in[12];
  const float* f1_b1   = (const float*)d_in[13];
  const float* f1_w2   = (const float*)d_in[14];
  const float* f1_b2   = (const float*)d_in[15];
  const float* f2_ln_g = (const float*)d_in[16];
  const float* f2_ln_b = (const float*)d_in[17];
  const float* f2_w1   = (const float*)d_in[18];
  const float* f2_b1   = (const float*)d_in[19];
  const float* f2_w2   = (const float*)d_in[20];
  const float* f2_b2   = (const float*)d_in[21];
  float* out = (float*)d_out;

  char* ws = (char*)d_ws;
  size_t off = 0;
  auto grab = [&](size_t bytes) -> void* {
    void* p = (void*)(ws + off);
    off = (off + bytes + 255) & ~(size_t)255;
    return p;
  };
  float*  zn      = (float*) grab((size_t)kB * kL * kD * 4);   // later: prod
  float*  cwhz    = (float*) grab((size_t)kB * kL * kD * 4);
  float2* wfh     = (float2*)grab((size_t)kD * kNF * 8);       // later: FFN hidden (bf16)
  __bf16* xln     = (__bf16*)grab((size_t)kMT * kD * 2);
  float*  hbuf    = (float*) grab((size_t)kL * kD * 4);
  __bf16* hid     = (__bf16*)grab((size_t)kL * kH1 * 2);
  __bf16* peb     = (__bf16*)grab((size_t)kL * kP * 2);
  __bf16* wb_lp1  = (__bf16*)grab((size_t)kP * kH1 * 2);       // [H1][P]
  __bf16* wb_lp2  = (__bf16*)grab((size_t)kH1 * kD * 2);       // [D][H1]
  __bf16* wb_f1w1 = (__bf16*)grab((size_t)kD * kDF * 2);       // [DF][D]
  __bf16* wb_f1w2 = (__bf16*)grab((size_t)kDF * kD * 2);       // [D][DF]
  __bf16* wb_f2w1 = (__bf16*)grab((size_t)kD * kDF * 2);
  __bf16* wb_f2w2 = (__bf16*)grab((size_t)kDF * kD * 2);
  __bf16* tbuf    = (__bf16*)wfh;      // FFN hidden reuses wfh after conv
  float*  prod    = zn;                // pre-LN product reuses zn after conv

  auto cvt = [&](const float* src, __bf16* dst, int n) {
    cvt_bf16_kernel<<<(n + 255) / 256, 256, 0, stream>>>(src, dst, n);
  };
  auto cvt_t = [&](const float* src, __bf16* dst, int K, int N) {
    int n = K * N;
    cvt_bf16_t_kernel<<<(n + 255) / 256, 256, 0, stream>>>(src, dst, K, N);
  };
  cvt(pe, peb, kL * kP);
  cvt_t(lp1_w, wb_lp1, kP, kH1);
  cvt_t(lp2_w, wb_lp2, kH1, kD);
  cvt_t(f1_w1, wb_f1w1, kD, kDF);
  cvt_t(f1_w2, wb_f1w2, kDF, kD);
  cvt_t(f2_w1, wb_f2w1, kD, kDF);
  cvt_t(f2_w2, wb_f2w2, kDF, kD);

  auto gemm = [&](const __bf16* A, const __bf16* Bt, const float* bias, void* C,
                  const float* r1, const float* r2, int M, int N, int K, int mode) {
    dim3 grid(N / TNg, M / TMg);
    gemm_bf16_kernel<<<grid, 256, 0, stream>>>(A, Bt, bias, C, r1, r2, M, N, K, mode);
  };

  // --- filter branch: h = silu(pe @ lp1_w + b1) @ lp2_w + b2 ---
  gemm(peb, wb_lp1, lp1_b, hid, nullptr, nullptr, kL, kH1, kP, 0);
  gemm(hid, wb_lp2, lp2_b, hbuf, nullptr, nullptr, kL, kD, kH1, 1);
  fft_filter_kernel<<<kD, 256, 2 * kNF * sizeof(float2), stream>>>(hbuf, a, wfh);

  // --- z branch: zn = LN(z); cwhz = irfft(wfh * rfft(zn))[:L] ---
  ln_kernel<<<kB * kL, 256, 0, stream>>>(z, nullptr, ln_g, ln_b, zn, nullptr, nullptr, kD);
  fft_conv_kernel<<<kB * kD, 256, 2 * kNF * sizeof(float2), stream>>>(zn, wfh, cwhz);

  // --- x branch: x1 = FFN1(x) + x  (x1 staged in d_out) ---
  ln_kernel<<<kB * kL, 256, 0, stream>>>(x, nullptr, f1_ln_g, f1_ln_b, nullptr, xln, nullptr, kD);
  gemm(xln, wb_f1w1, f1_b1, tbuf, nullptr, nullptr, kMT, kDF, kD, 0);
  gemm(tbuf, wb_f1w2, f1_b2, out, x, nullptr, kMT, kD, kDF, 1);

  // --- gate + FFN2: y = x1*cwhz; out = FFN2(y) + y + z ---
  ln_kernel<<<kB * kL, 256, 0, stream>>>(out, cwhz, f2_ln_g, f2_ln_b, nullptr, xln, prod, kD);
  gemm(xln, wb_f2w1, f2_b1, tbuf, nullptr, nullptr, kMT, kDF, kD, 0);
  gemm(tbuf, wb_f2w2, f2_b2, out, prod, z, kMT, kD, kDF, 1);
}
